// Self_Attn_77592879169813
// MI455X (gfx1250) — matrix operand order repaired and checked by reading, not compiled, not run on hardware
//
#include <hip/hip_runtime.h>
#include <stdint.h>

#define BB  4
#define CC  256
#define HHH 64
#define WWW 64
#define NN  4096
#define DD  32
#define MIPC 8
#define KCH 64            // attention key-chunk
#define PAD 8

typedef __attribute__((ext_vector_type(16))) _Float16 v16h;
typedef __attribute__((ext_vector_type(8)))  float    v8f;
typedef int v4i_vs __attribute__((__vector_size__(16)));       // builtin param style
typedef unsigned int v4u_vs __attribute__((__vector_size__(16)));
typedef int v8i_vs __attribute__((__vector_size__(32)));

union ABu { v16h v; _Float16 h[16]; uint32_t u[8]; uint4 q[2]; };

__device__ __forceinline__ v8f wmma16(const ABu& a, const ABu& b, v8f c) {
  // 8 args: (neg_a, A, neg_b, B, c_mod, C, reuse_a, reuse_b)
  return __builtin_amdgcn_wmma_f32_16x16x32_f16(false, a.v, false, b.v,
                                                (short)0, c, false, false);
}

// ---- gfx1250 async global->LDS path (probe-confirmed on this toolchain) ----
#if defined(__has_builtin)
#  if __has_builtin(__builtin_amdgcn_global_load_async_to_lds_b128)
#    define HAVE_ASYNC_LDS 1
#  endif
#  if __has_builtin(__builtin_amdgcn_tensor_load_to_lds)
#    define HAVE_TDM 1
#  endif
#endif
#ifndef HAVE_ASYNC_LDS
#  define HAVE_ASYNC_LDS 0
#endif
#ifndef HAVE_TDM
#  define HAVE_TDM 0
#endif

#define GPTR128(p) ((__attribute__((address_space(1))) v4i_vs*)(uintptr_t)(p))
#define LPTR128(p) ((__attribute__((address_space(3))) v4i_vs*)(uint32_t)(uintptr_t)(p))

__device__ __forceinline__ void async_cp16(const _Float16* g, _Float16* l) {
#if HAVE_ASYNC_LDS
  __builtin_amdgcn_global_load_async_to_lds_b128(GPTR128(g), LPTR128(l), 0, 0);
#else
  *(uint4*)l = *(const uint4*)g;
#endif
}

__device__ __forceinline__ void wait_async_all() {
#if HAVE_ASYNC_LDS
#  if __has_builtin(__builtin_amdgcn_s_wait_asynccnt)
  __builtin_amdgcn_s_wait_asynccnt(0);
#  else
  asm volatile("s_wait_asynccnt 0x0" ::: "memory");
#  endif
#endif
}

__device__ __forceinline__ void wait_tensor() {
#if HAVE_TDM
#  if __has_builtin(__builtin_amdgcn_s_wait_tensorcnt)
  __builtin_amdgcn_s_wait_tensorcnt(0);
#  else
  asm volatile("s_wait_tensorcnt 0x0" ::: "memory");
#  endif
#endif
}

// TDM 2-D tile load: d1 rows of d0 f16 elements, row stride `stride` elements
// in memory, +16B LDS padding after each row (pad_amount code 3 = 4 DWORDs).
// padcode: pad_interval code s.t. (row bytes / 4) DWORDs = table value.
__device__ __forceinline__ void tdm_load_2d(const _Float16* g, _Float16* lds,
                                            uint32_t d0, uint32_t d1,
                                            uint32_t stride, uint32_t padcode) {
#if HAVE_TDM
  uint64_t ga = (uint64_t)(uintptr_t)g;       // byte address (57-bit field)
  uint32_t la = (uint32_t)(uintptr_t)lds;     // LDS byte offset
  v4u_vs g0 = { 1u,                                     // count=1, user desc
                la,                                     // lds_addr
                (uint32_t)ga,                           // global_addr[31:0]
                ((uint32_t)(ga >> 32) & 0x01FFFFFFu) | (2u << 30) }; // [56:32]|type=2
  v8i_vs g1 = {
    (int)((1u << 16) | (1u << 20) | (padcode << 22) | (3u << 25)),
    //    data_size=2B  pad_enable   pad_interval       pad_amount=4 DWORDs
    (int)(d0 << 16),                 // [63:48]  tensor_dim0[15:0]
    (int)(d1 << 16),                 // [95:80]  tensor_dim1[15:0]
    (int)(d0 << 16),                 // [127:112] tile_dim0
    (int)(d1),                       // [143:128] tile_dim1 ([159:144] tile_dim2=0)
    (int)(stride),                   // [191:160] tensor_dim0_stride[31:0]
    0, 0 };
  v4i_vs gz = { 0, 0, 0, 0 };        // groups 2/3 unused (2-D tensor)
#  if __clang_major__ >= 23
  v8i_vs gz8 = { 0, 0, 0, 0, 0, 0, 0, 0 };
  __builtin_amdgcn_tensor_load_to_lds(g0, g1, gz, gz, gz8, 0);
#  else
  __builtin_amdgcn_tensor_load_to_lds(g0, g1, gz, gz, 0);
#  endif
#else
  (void)g; (void)lds; (void)d0; (void)d1; (void)stride; (void)padcode;
#endif
}

// ---------------------------------------------------------------------------
// Kernel 0a: elementwise f32 -> f16 (weights)
// ---------------------------------------------------------------------------
__global__ void cvt16_kernel(const float* __restrict__ src,
                             _Float16* __restrict__ dst, int count) {
  int i = blockIdx.x * 256 + threadIdx.x;
  if (i < count) dst[i] = (_Float16)src[i];
}

// ---------------------------------------------------------------------------
// Kernel 0b: x (B,C,N) f32 -> xT16 (B,N,C) f16, LDS-tiled transpose
// ---------------------------------------------------------------------------
__global__ __launch_bounds__(256)
void xpose_kernel(const float* __restrict__ x, _Float16* __restrict__ xT) {
  __shared__ _Float16 tile[64][64 + PAD];
  int b = blockIdx.z, c0 = blockIdx.y * 64, n0 = blockIdx.x * 64;
  int tn = threadIdx.x & 63, tg = threadIdx.x >> 6;
#pragma unroll
  for (int i = 0; i < 16; ++i) {
    int cl = tg * 16 + i;
    tile[cl][tn] = (_Float16)x[((size_t)(b * CC + c0 + cl)) * NN + n0 + tn];
  }
  __syncthreads();
#pragma unroll
  for (int i = 0; i < 16; ++i) {
    int nl = tg * 16 + i;
    xT[((size_t)(b * NN + n0 + nl)) * CC + c0 + tn] = tile[tn][nl];
  }
}

// ---------------------------------------------------------------------------
// Kernel 1: per-(b,c) row means (over W) and column means (over H)
// ---------------------------------------------------------------------------
__global__ void ca_mean_kernel(const float* __restrict__ x,
                               float* __restrict__ xh, float* __restrict__ xw) {
  int bc = blockIdx.x;          // b*C + c
  int t  = threadIdx.x;         // 0..63
  const float* p = x + (size_t)bc * (HHH * WWW);
  float rs = 0.f, cs = 0.f;
#pragma unroll 4
  for (int i = 0; i < WWW; ++i) {
    rs += p[t * WWW + i];       // row t, mean over w  -> x_h[b,c,t]
    cs += p[i * WWW + t];       // col t, mean over h  -> x_w[b,c,t]
  }
  xh[bc * HHH + t] = rs * (1.f / WWW);
  xw[bc * WWW + t] = cs * (1.f / HHH);
}

// ---------------------------------------------------------------------------
// Kernel 2: coord-att MLP -> a_h[b,d,h], a_w[b,d,w]
// ---------------------------------------------------------------------------
__global__ void ca_mlp_kernel(const float* __restrict__ xh, const float* __restrict__ xw,
                              const float* __restrict__ c1w, const float* __restrict__ c1b,
                              const float* __restrict__ bng, const float* __restrict__ bnb,
                              const float* __restrict__ hwt, const float* __restrict__ hbt,
                              const float* __restrict__ wwt, const float* __restrict__ wbt,
                              float* __restrict__ ah, float* __restrict__ aw) {
  int b  = blockIdx.x;
  int t  = threadIdx.x;                 // 0..127
  int isW = (t >= HHH);
  int idx = isW ? (t - HHH) : t;
  const float* src = isW ? (xw + (size_t)(b * CC) * WWW + idx)
                         : (xh + (size_t)(b * CC) * HHH + idx);
  float y[MIPC];
#pragma unroll
  for (int j = 0; j < MIPC; ++j) y[j] = 0.f;
  for (int c = 0; c < CC; ++c) {
    float v = src[c * HHH];
#pragma unroll
    for (int j = 0; j < MIPC; ++j) y[j] += c1w[j * CC + c] * v;
  }
  const float invbn = 1.0f / sqrtf(1.0f + 1e-5f);
#pragma unroll
  for (int j = 0; j < MIPC; ++j) {
    float v  = (y[j] + c1b[j]) * invbn * bng[j] + bnb[j];
    float cl = fminf(fmaxf(v + 3.f, 0.f), 6.f);
    y[j] = v * cl * (1.f / 6.f);        // h_swish
  }
  const float* W2 = isW ? wwt : hwt;
  const float* B2 = isW ? wbt : hbt;
  float* dst = isW ? aw : ah;
  for (int d = 0; d < DD; ++d) {
    float o = B2[d];
#pragma unroll
    for (int j = 0; j < MIPC; ++j) o += W2[d * MIPC + j] * y[j];
    dst[(b * DD + d) * HHH + idx] = 1.f / (1.f + __expf(-o));
  }
}

// ---------------------------------------------------------------------------
// Kernel 3: WMMA projections (all-f16 operands, zero converts in hot loop).
// qh:(B,N,D) f16 (x0.5 folded), khT:(B,N,D) f16 (K-bias fused), vh:(B,C,N) f16
// ---------------------------------------------------------------------------
#define XS_STR (CC + PAD)   // 264 f16 per row (512B + 16B TDM pad)

__device__ __forceinline__ ABu loadA16(const _Float16* __restrict__ W,
                                       int row, int c0, int half) {
  ABu a;
  const _Float16* p = W + row * CC + c0 + half * 16;
  a.q[0] = *(const uint4*)p;          // K run 0
  a.q[1] = *(const uint4*)(p + 8);    // K run 1
  return a;
}

__device__ __forceinline__ ABu loadB_lds(const _Float16* __restrict__ row,
                                         int kbase) {
  ABu bt;
  const uint32_t* p = (const uint32_t*)(row + kbase);
#pragma unroll
  for (int i = 0; i < 8; ++i) bt.u[i] = p[i];
  return bt;
}

__global__ __launch_bounds__(128)
void proj_kernel(const _Float16* __restrict__ xT,
                 const _Float16* __restrict__ qW, const float* __restrict__ qB,
                 const _Float16* __restrict__ kW, const float* __restrict__ kB,
                 const _Float16* __restrict__ vW, const float* __restrict__ vB,
                 const float* __restrict__ ah, const float* __restrict__ aw,
                 _Float16* __restrict__ qh, _Float16* __restrict__ khT,
                 _Float16* __restrict__ vh) {
  __shared__ _Float16 Xs[64][XS_STR];   // [n_local][c], c contiguous
  int b  = blockIdx.y;
  int n0 = blockIdx.x * 64;
  int tid = threadIdx.x;
  int wave = tid >> 5, lane = tid & 31;
#if HAVE_TDM
  if (wave == 0)     // one TDM op DMAs the whole 64x256 f16 tile (+row pad)
    tdm_load_2d(xT + ((size_t)(b * NN + n0)) * CC, &Xs[0][0],
                CC, 64, CC, /*512B row = 128 DW*/ 6);
  wait_tensor();
#else
#pragma unroll
  for (int it = 0; it < 16; ++it) {
    int idx = it * 128 + tid;
    int row = idx >> 5, qi = idx & 31;
    async_cp16(xT + ((size_t)(b * NN + n0 + row)) * CC + qi * 8,
               &Xs[row][qi * 8]);
  }
  wait_async_all();
#endif
  __syncthreads();

  int mcol = lane & 15, half = lane >> 4;
  int ncol = n0 + wave * 16 + mcol;                 // this lane's column (n)
  const _Float16* xrow = &Xs[wave * 16 + mcol][0];
  const v8f zacc = {};

  // ---- Q (2 tiles of 16 out-channels) ----
#pragma unroll
  for (int t = 0; t < 2; ++t) {
    int o0 = t * 16;
    v8f acc = zacc;
    for (int c0 = 0; c0 < CC; c0 += 32) {
      ABu a  = loadA16(qW, o0 + mcol, c0, half);
      ABu bt = loadB_lds(xrow, c0 + half * 16);
      acc = wmma16(a, bt, acc);
    }
#pragma unroll
    for (int r = 0; r < 8; ++r) {
      int d = o0 + r + 8 * half;
      float val = (acc[r] + qB[d]) * 0.5f;          // fold 1/sqrt(B)
      qh[((size_t)(b * NN + ncol)) * DD + d] = (_Float16)val;
    }
  }

  // ---- K (2 tiles), with coord-att bias fused ----
  int hh = ncol >> 6, wi = ncol & 63;
#pragma unroll
  for (int t = 0; t < 2; ++t) {
    int o0 = t * 16;
    v8f acc = zacc;
    for (int c0 = 0; c0 < CC; c0 += 32) {
      ABu a  = loadA16(kW, o0 + mcol, c0, half);
      ABu bt = loadB_lds(xrow, c0 + half * 16);
      acc = wmma16(a, bt, acc);
    }
#pragma unroll
    for (int r = 0; r < 8; ++r) {
      int d = o0 + r + 8 * half;
      float val = acc[r] + kB[d];
      float xv = (float)xT[((size_t)(b * NN + ncol)) * CC + d];
      val += ah[(b * DD + d) * HHH + hh] * aw[(b * DD + d) * WWW + wi] * xv;
      khT[((size_t)(b * NN + ncol)) * DD + d] = (_Float16)val;
    }
  }

  // ---- V (16 tiles) ----
  for (int t = 0; t < 16; ++t) {
    int o0 = t * 16;
    v8f acc = zacc;
    for (int c0 = 0; c0 < CC; c0 += 32) {
      ABu a  = loadA16(vW, o0 + mcol, c0, half);
      ABu bt = loadB_lds(xrow, c0 + half * 16);
      acc = wmma16(a, bt, acc);
    }
#pragma unroll
    for (int r = 0; r < 8; ++r) {
      int c = o0 + r + 8 * half;
      vh[((size_t)(b * CC + c)) * NN + ncol] = (_Float16)(acc[r] + vB[c]);
    }
  }
}

// ---------------------------------------------------------------------------
// Kernel 4: fused attention, double-buffered TDM/async LDS pipeline
// ---------------------------------------------------------------------------
#define KSTR (DD + PAD)            // 40 f16  (KsT row stride: 64B + 16B pad)
#define VSTR (KCH + PAD)           // 72 f16  (Vs/Ps row stride: 128B + 16B pad)
#define KS_BYTES (KCH * KSTR * 2)  // 5120
#define VS_BYTES (CC * VSTR * 2)   // 36864
#define PS_BYTES (4 * 16 * VSTR * 2)
#define ATTN_SMEM (2 * KS_BYTES + 2 * VS_BYTES + PS_BYTES)   // 93184 B

__device__ __forceinline__ void stage_chunk(const _Float16* __restrict__ khT,
                                            const _Float16* __restrict__ vh,
                                            _Float16* KsT, _Float16* Vs,
                                            int b, int j0, int tid) {
  int i0 = tid * 2;                                   // KsT: 256 x 16B
#pragma unroll
  for (int i = i0; i < i0 + 2; ++i) {
    int key = i >> 2, qi = i & 3;
    async_cp16(khT + ((size_t)(b * NN + j0 + key)) * DD + qi * 8,
               KsT + key * KSTR + qi * 8);
  }
#pragma unroll
  for (int it = 0; it < 16; ++it) {                   // Vs: 2048 x 16B
    int idx = it * 128 + tid;
    int row = idx >> 3, qi = idx & 7;
    async_cp16(vh + ((size_t)(b * CC + row)) * NN + j0 + qi * 8,
               Vs + row * VSTR + qi * 8);
  }
}

__device__ __forceinline__ void stage_tdm(const _Float16* __restrict__ khT,
                                          const _Float16* __restrict__ vh,
                                          _Float16* KsT, _Float16* Vs,
                                          int b, int j0) {
  // K chunk: 64 rows x 32 f16 (64B row = 16 DW -> padcode 3), contiguous rows
  tdm_load_2d(khT + ((size_t)(b * NN + j0)) * DD, KsT, DD, KCH, DD, 3);
  // V chunk: 256 rows x 64 f16 (128B row = 32 DW -> padcode 4), stride N
  tdm_load_2d(vh + ((size_t)(b * CC)) * NN + j0, Vs, KCH, CC, NN, 4);
}

__global__ __launch_bounds__(128)
void attn_kernel(const float* __restrict__ x, const float* __restrict__ gamma,
                 const _Float16* __restrict__ qh, const _Float16* __restrict__ khT,
                 const _Float16* __restrict__ vh, float* __restrict__ out) {
  extern __shared__ char smem[];
  _Float16* Ks0 = (_Float16*)(smem);
  _Float16* Ks1 = (_Float16*)(smem + KS_BYTES);
  _Float16* Vs0 = (_Float16*)(smem + 2 * KS_BYTES);
  _Float16* Vs1 = (_Float16*)(smem + 2 * KS_BYTES + VS_BYTES);
  _Float16* Psb = (_Float16*)(smem + 2 * KS_BYTES + 2 * VS_BYTES);

  int b   = blockIdx.y;
  int tid = threadIdx.x, wave = tid >> 5, lane = tid & 31;
  int mcol = lane & 15, half = lane >> 4;
  int m0 = (blockIdx.x * 4 + wave) * 16;   // query tile base
  _Float16* Psw = Psb + wave * 16 * VSTR;  // per-wave P scratch

  // Q tile in A layout
  ABu qa;
  {
    const _Float16* qp = qh + ((size_t)(b * NN + m0 + mcol)) * DD;
    qa.q[0] = *(const uint4*)(qp + half * 16);
    qa.q[1] = *(const uint4*)(qp + half * 16 + 8);
  }

  const v8f zacc = {};
  v8f O[16];
#pragma unroll
  for (int i = 0; i < 16; ++i) O[i] = zacc;
  float lsum[8];
#pragma unroll
  for (int r = 0; r < 8; ++r) lsum[r] = 0.f;

#if HAVE_TDM
  if (wave == 0) stage_tdm(khT, vh, Ks0, Vs0, b, 0);
#else
  stage_chunk(khT, vh, Ks0, Vs0, b, 0, tid);
#endif

  for (int j0 = 0; j0 < NN; j0 += KCH) {
    int cur = (j0 / KCH) & 1;
    _Float16* Ksc = cur ? Ks1 : Ks0;
    _Float16* Vsc = cur ? Vs1 : Vs0;

#if HAVE_TDM
    wait_tensor();           // wave 0: its TDM tiles are in LDS (others cnt=0)
#else
    wait_async_all();        // our staged chunk is in LDS
#endif
    __syncthreads();         // everyone sees the chunk; prev compute done

    if (j0 + KCH < NN) {     // prefetch next chunk into the other buffer
#if HAVE_TDM
      if (wave == 0)
        stage_tdm(khT, vh, cur ? Ks0 : Ks1, cur ? Vs0 : Vs1, b, j0 + KCH);
#else
      stage_chunk(khT, vh, cur ? Ks0 : Ks1, cur ? Vs0 : Vs1, b, j0 + KCH, tid);
#endif
    }

    // ---- S = Q*K, P = exp(S), row-partials into lsum, P -> LDS ----
#pragma unroll
    for (int t = 0; t < KCH / 16; ++t) {
      ABu kb;
      const uint32_t* p =
          (const uint32_t*)(Ksc + (t * 16 + mcol) * KSTR + half * 16);
#pragma unroll
      for (int i = 0; i < 8; ++i) kb.u[i] = p[i];
      v8f s = wmma16(qa, kb, zacc);
#pragma unroll
      for (int r = 0; r < 8; ++r) {
        float pv = __expf(s[r]);   // logits are O(1): exact softmax w/o max
        lsum[r] += pv;
        Psw[(r + 8 * half) * VSTR + t * 16 + mcol] = (_Float16)pv;
      }
    }

    // ---- O += P * V^T ----
#pragma unroll
    for (int kk = 0; kk < KCH / 32; ++kk) {
      ABu pa;
      const _Float16* pr = Psw + mcol * VSTR + kk * 32;
      pa.q[0] = *(const uint4*)(pr + half * 16);
      pa.q[1] = *(const uint4*)(pr + half * 16 + 8);
#pragma unroll
      for (int ct = 0; ct < 16; ++ct) {
        ABu vb;
        const uint32_t* vp =
            (const uint32_t*)(Vsc + (ct * 16 + mcol) * VSTR + kk * 32 + half * 16);
#pragma unroll
        for (int i = 0; i < 8; ++i) vb.u[i] = vp[i];
        O[ct] = wmma16(pa, vb, O[ct]);
      }
    }
  }

  // ---- reduce lsum across the 16 lanes holding one row, then epilogue ----
#pragma unroll
  for (int r = 0; r < 8; ++r) {
    float v = lsum[r];
    v += __shfl_xor(v, 1, 32);
    v += __shfl_xor(v, 2, 32);
    v += __shfl_xor(v, 4, 32);
    v += __shfl_xor(v, 8, 32);
    lsum[r] = 1.0f / v;
  }
  float g = gamma[0];
#pragma unroll
  for (int ct = 0; ct < 16; ++ct) {
    int c = ct * 16 + mcol;
#pragma unroll
    for (int r = 0; r < 8; ++r) {
      int n = m0 + r + 8 * half;
      size_t off = ((size_t)(b * CC + c)) * NN + n;
      out[off] = g * O[ct][r] * lsum[r] + x[off];
    }
  }
}

// ---------------------------------------------------------------------------
extern "C" void kernel_launch(void* const* d_in, const int* in_sizes, int n_in,
                              void* d_out, int out_size, void* d_ws, size_t ws_size,
                              hipStream_t stream) {
  (void)in_sizes; (void)n_in; (void)out_size; (void)ws_size;
  const float* x    = (const float*)d_in[0];
  const float* q_w  = (const float*)d_in[1];
  const float* q_b  = (const float*)d_in[2];
  const float* k_w  = (const float*)d_in[3];
  const float* k_b  = (const float*)d_in[4];
  const float* v_w  = (const float*)d_in[5];
  const float* v_b  = (const float*)d_in[6];
  const float* c1w  = (const float*)d_in[7];
  const float* c1b  = (const float*)d_in[8];
  const float* bng  = (const float*)d_in[9];
  const float* bnb  = (const float*)d_in[10];
  const float* hw   = (const float*)d_in[11];
  const float* hb   = (const float*)d_in[12];
  const float* ww_  = (const float*)d_in[13];
  const float* wb   = (const float*)d_in[14];
  const float* gam  = (const float*)d_in[15];
  float* out = (float*)d_out;

  char* ws = (char*)d_ws;
  const size_t MB = 1u << 20;
  _Float16* qh   = (_Float16*)(ws + 0 * MB);         // 1 MiB  (B*N*D)
  _Float16* khT  = (_Float16*)(ws + 1 * MB);         // 1 MiB
  _Float16* vh   = (_Float16*)(ws + 2 * MB);         // 8 MiB  (B*C*N)
  _Float16* xT16 = (_Float16*)(ws + 10 * MB);        // 8 MiB  (B,N,C) f16
  _Float16* wq16 = (_Float16*)(ws + 18 * MB);        // 16 KiB (D*C)
  _Float16* wk16 = (_Float16*)(ws + 18 * MB + 0x4000);
  _Float16* wv16 = (_Float16*)(ws + 18 * MB + 0x8000);   // 128 KiB (C*C)
  float* xh = (float*)(ws + 18 * MB + 0x28000);      // 256 KiB
  float* xw = (float*)(ws + 18 * MB + 0x68000);      // 256 KiB
  float* ah = (float*)(ws + 18 * MB + 0xA8000);      // 32 KiB
  float* aw = (float*)(ws + 18 * MB + 0xB0000);      // 32 KiB

  cvt16_kernel<<<dim3((DD * CC + 255) / 256), dim3(256), 0, stream>>>(q_w, wq16, DD * CC);
  cvt16_kernel<<<dim3((DD * CC + 255) / 256), dim3(256), 0, stream>>>(k_w, wk16, DD * CC);
  cvt16_kernel<<<dim3((CC * CC + 255) / 256), dim3(256), 0, stream>>>(v_w, wv16, CC * CC);
  xpose_kernel<<<dim3(NN / 64, CC / 64, BB), dim3(256), 0, stream>>>(x, xT16);
  ca_mean_kernel<<<dim3(BB * CC), dim3(64), 0, stream>>>(x, xh, xw);
  ca_mlp_kernel<<<dim3(BB), dim3(128), 0, stream>>>(xh, xw, c1w, c1b, bng, bnb,
                                                    hw, hb, ww_, wb, ah, aw);
  proj_kernel<<<dim3(NN / 64, BB), dim3(128), 0, stream>>>(
      xT16, wq16, q_b, wk16, k_b, wv16, v_b, ah, aw, qh, khT, vh);
  attn_kernel<<<dim3(NN / 64, BB), dim3(128), ATTN_SMEM, stream>>>(
      x, gam, qh, khT, vh, out);
}
